// CGConvLayer_47974784696405
// MI455X (gfx1250) — matrix operand order repaired and checked by reading, not compiled, not run on hardware
//
#include <hip/hip_runtime.h>
#include <hip/hip_bf16.h>

typedef __attribute__((ext_vector_type(16))) __bf16 v16bf;
typedef __attribute__((ext_vector_type(8)))  float  v8f;

#define OUT_CH  128
#define EDGE_CH 64
#define K_TOT   192            // OUT_CH + EDGE_CH
#define KH_TOT  (K_TOT / 2)    // 96 packed bf16 pairs per column
#define KH_PAD  100            // padded column stride (u32) -> conflict-free LDS
#define WQ_U32  (OUT_CH * KH_PAD)      // 12800 u32 = 51.2 KB padded weight image
#define WQ_V4   (WQ_U32 / 4)           // 3200 uint4 tiles to stage

// f32 -> bf16 (round half up) pair packed into one u32 via a single v_perm_b32
static __device__ __forceinline__ unsigned int pack_bf16(float a, float b) {
    unsigned int ua = __float_as_uint(a) + 0x8000u;
    unsigned int ub = __float_as_uint(b) + 0x8000u;
    return __builtin_amdgcn_perm(ub, ua, 0x07060302u);
}

union Frag16 {
    v16bf v;
    unsigned int u[8];
    uint4 q[2];
};

__global__ void __launch_bounds__(256)
cg_zero_kernel(float* __restrict__ out, int total, float* __restrict__ cnt, int n_nodes) {
    int i = blockIdx.x * blockDim.x + threadIdx.x;
    if (i < total) out[i] = 0.0f;
    if (i < n_nodes) cnt[i] = 0.0f;
}

__global__ void __launch_bounds__(256)
cg_count_kernel(const int* __restrict__ row, float* __restrict__ cnt, int n_edges) {
    int i = blockIdx.x * blockDim.x + threadIdx.x;
    if (i < n_edges) atomicAdd(&cnt[row[i]], 1.0f);
}

// One-time weight conversion: f32 W[K][N] row-major -> packed bf16,
// column-major along K with padded stride:
//   wq[n*KH_PAD + k/2] = { W[k][n], W[k+1][n] },  pad entries = 0
__global__ void __launch_bounds__(256)
cg_wcvt_kernel(const float* __restrict__ weight, unsigned int* __restrict__ wq) {
    int i = blockIdx.x * blockDim.x + threadIdx.x;
    if (i >= WQ_U32) return;
    int n  = i / KH_PAD;
    int kh = i - n * KH_PAD;
    unsigned int val = 0u;
    if (kh < KH_TOT) {
        int k = kh * 2;
        val = pack_bf16(weight[k * OUT_CH + n], weight[(k + 1) * OUT_CH + n]);
    }
    wq[i] = val;
}

// One wave = one 16-edge tile.  D(16x128) = A(16x192) * W(192x128), bf16 WMMA,
// f32 accumulate, then scatter-add rows into the node accumulator (d_out).
__global__ void __launch_bounds__(256)
cg_msg_wmma_kernel(const float* __restrict__ x,
                   const int*   __restrict__ row,
                   const int*   __restrict__ col,
                   const float* __restrict__ edge_attr,
                   const unsigned int* __restrict__ wq,   // padded bf16 weight
                   float*       __restrict__ accum,
                   int n_edges, int n_tiles) {
    // bf16 weight in LDS, padded column stride (51.2 KB)
    __shared__ unsigned int wlds[WQ_U32];

    // Stage via CDNA5 async global->LDS DMA (GVS mode), 16B per lane per issue.
    {
        unsigned int lds_base = (unsigned int)(size_t)(&wlds[0]);
#pragma unroll
        for (int it = 0; it < (WQ_V4 + 255) / 256; ++it) {   // 13 iters, last partial
            int idx = threadIdx.x + it * 256;
            if (idx < WQ_V4) {
                unsigned int byte_off = (unsigned int)idx * 16u;
                unsigned int lds_addr = lds_base + byte_off;
                asm volatile("global_load_async_to_lds_b128 %0, %1, %2"
                             :
                             : "v"(lds_addr), "v"(byte_off), "s"(wq)
                             : "memory");
            }
        }
        asm volatile("s_wait_asynccnt 0" ::: "memory");
    }
    __syncthreads();

    // scalarize the tile id so all tile-level control flow is SALU
    const int tile = __builtin_amdgcn_readfirstlane(blockIdx.x * 8 + (threadIdx.x >> 5));
    if (tile >= n_tiles) return;

    const int tbase = tile * 16;                        // SGPR
    const bool tile_full = (tbase + 16 <= n_edges);     // SGPR predicate
    const int lane  = threadIdx.x & 31;
    const int m_lo  = lane & 15;   // A row / D column-lane
    const int khalf = lane >> 4;   // 0: lanes 0-15, 1: lanes 16-31

    // --- A fragments: row m = tbase + m_lo, K = 0..191 (x[col] ++ edge_attr) ---
    int e_a = tbase + m_lo;
    if (e_a >= n_edges) e_a = n_edges - 1;              // clamp for safe loads
    const int   cidx = col[e_a];
    const float* xr  = x + (size_t)cidx * OUT_CH;
    const float* ar  = edge_attr + (size_t)e_a * EDGE_CH;

    Frag16 a[6];
#pragma unroll
    for (int ks = 0; ks < 6; ++ks) {
        // ks 0..3 read x (K<128), ks 4..5 read edge_attr (K>=128); 128 % 32 == 0
        const float* src = (ks < 4) ? (xr + ks * 32) : (ar + (ks - 4) * 32);
        // ISA 16-bit A layout: lane-lo K {0..7,16..23}, lane-hi K {8..15,24..31}
        const float4* p0 = (const float4*)(src + khalf * 8);        // K base + 0
        const float4* p1 = (const float4*)(src + 16 + khalf * 8);   // K base + 16
        float4 f0 = p0[0], f1 = p0[1];
        float4 f2 = p1[0], f3 = p1[1];
        a[ks].u[0] = pack_bf16(f0.x, f0.y);
        a[ks].u[1] = pack_bf16(f0.z, f0.w);
        a[ks].u[2] = pack_bf16(f1.x, f1.y);
        a[ks].u[3] = pack_bf16(f1.z, f1.w);
        a[ks].u[4] = pack_bf16(f2.x, f2.y);
        a[ks].u[5] = pack_bf16(f2.z, f2.w);
        a[ks].u[6] = pack_bf16(f3.x, f3.y);
        a[ks].u[7] = pack_bf16(f3.z, f3.w);
    }

    // Ragged last tile only (scalar branch, never taken when 16 | n_edges):
    // zero the A rows of out-of-range edges so their D rows are exactly 0 and
    // the unguarded atomicAdd below adds 0.0f (a no-op) to a clamped node.
    if (!tile_full) {
        unsigned int keep = ((tbase + m_lo) < n_edges) ? ~0u : 0u;
#pragma unroll
        for (int ks = 0; ks < 6; ++ks)
#pragma unroll
            for (int v = 0; v < 8; ++v)
                a[ks].u[v] &= keep;
    }

    // Destination nodes for this lane's 8 D rows (m = khalf*8 + v), clamped
    int doff[8];
#pragma unroll
    for (int v = 0; v < 8; ++v) {
        int e = tbase + (khalf << 3) + v;
        if (e >= n_edges) e = n_edges - 1;
        doff[v] = row[e] << 7;   // node * OUT_CH
    }

    const int ncol_lane = m_lo;
#pragma unroll
    for (int nt = 0; nt < 8; ++nt) {
        const int ncol = nt * 16 + ncol_lane;
        v8f c = {0.f, 0.f, 0.f, 0.f, 0.f, 0.f, 0.f, 0.f};
#pragma unroll
        for (int ks = 0; ks < 6; ++ks) {
            // B 32x16 bf16 layout: lane-lo K {0..7,16..23}, lane-hi K {8..15,24..31} (matches A)
            Frag16 b;
            const uint4* bp =
                (const uint4*)&wlds[ncol * KH_PAD + ks * 16 + khalf * 4];
            b.q[0] = bp[0];
            b.q[1] = bp[2];
            c = __builtin_amdgcn_wmma_f32_16x16x32_bf16(
                    /*neg_a=*/false, a[ks].v, /*neg_b=*/false, b.v,
                    /*c_mod=*/(short)0, c, /*reuse_a=*/false, /*reuse_b=*/false);
        }
        // D layout: VGPR v holds (M = v + 8*khalf, N = ncol) -> scatter-add
#pragma unroll
        for (int v = 0; v < 8; ++v)
            atomicAdd(&accum[doff[v] + ncol], c[v]);
    }
}

__global__ void __launch_bounds__(256)
cg_finalize_kernel(float* __restrict__ out, const float* __restrict__ cnt,
                   const float* __restrict__ bias, int total) {
    int i = blockIdx.x * blockDim.x + threadIdx.x;
    if (i >= total) return;
    int n = i >> 7;       // / OUT_CH
    int ch = i & 127;     // % OUT_CH
    float cv = cnt[n];
    cv = (cv < 1.0f) ? 1.0f : cv;
    out[i] = out[i] / cv + bias[ch];
}

extern "C" void kernel_launch(void* const* d_in, const int* in_sizes, int n_in,
                              void* d_out, int out_size, void* d_ws, size_t ws_size,
                              hipStream_t stream) {
    const float* x          = (const float*)d_in[0];
    const int*   edge_index = (const int*)  d_in[1];
    const float* edge_attr  = (const float*)d_in[2];
    const float* weight     = (const float*)d_in[3];
    const float* bias       = (const float*)d_in[4];

    const int n_nodes = in_sizes[0] / OUT_CH;
    const int n_edges = in_sizes[1] / 2;
    const int* row = edge_index;            // edge_index[0, :]
    const int* col = edge_index + n_edges;  // edge_index[1, :]

    float* out = (float*)d_out;
    float* cnt = (float*)d_ws;                                     // n_nodes f32
    unsigned int* wq = (unsigned int*)((char*)d_ws + (256 << 10)); // padded bf16 weight

    const int total = n_nodes * OUT_CH;

    cg_zero_kernel<<<(total + 255) / 256, 256, 0, stream>>>(out, total, cnt, n_nodes);
    cg_count_kernel<<<(n_edges + 255) / 256, 256, 0, stream>>>(row, cnt, n_edges);
    cg_wcvt_kernel<<<(WQ_U32 + 255) / 256, 256, 0, stream>>>(weight, wq);

    const int n_tiles = (n_edges + 15) / 16;
    cg_msg_wmma_kernel<<<(n_tiles + 7) / 8, 256, 0, stream>>>(
        x, row, col, edge_attr, wq, out, n_edges, n_tiles);

    cg_finalize_kernel<<<(total + 255) / 256, 256, 0, stream>>>(out, cnt, bias, total);
}
